// MultiHeadSelfAttention_60705067761672
// MI455X (gfx1250) — matrix operand order repaired and checked by reading, not compiled, not run on hardware
//
#include <hip/hip_runtime.h>
#include <hip/hip_bf16.h>

// ---------------------------------------------------------------------------
// MI455X (gfx1250) causal MHSA, WMMA-everywhere pipeline.
//   S=4096, D=1024, H=16, dk=64.  ~103 GFLOP vs ~100MB traffic -> compute
//   bound on the matrix pipes; all GEMMs run on v_wmma_f32_16x16x32_f16.
//   This revision software-pipelines the GEMM (ping-pong register buffers)
//   and hoists V-tile loads above the softmax in the attention kernel so the
//   compiler can emit partial s_wait_loadcnt instead of draining to zero.
// ---------------------------------------------------------------------------

typedef __attribute__((ext_vector_type(16))) _Float16 v16h;
typedef __attribute__((ext_vector_type(8)))  _Float16 v8h;
typedef __attribute__((ext_vector_type(4)))  _Float16 v4h;
typedef __attribute__((ext_vector_type(8)))  float    v8f;

#define SEQ    4096
#define DMODEL 1024
#define NHEADS 16
#define DK     64
#define NEG_BIG (-1.0e30f)

__device__ __forceinline__ v16h join16(v8h lo, v8h hi) {
  v16h r;
#pragma unroll
  for (int i = 0; i < 8; ++i) { r[i] = lo[i]; r[i + 8] = hi[i]; }
  return r;
}

__device__ __forceinline__ v8f vzero8() {
  v8f z;
#pragma unroll
  for (int i = 0; i < 8; ++i) z[i] = 0.0f;
  return z;
}

__device__ __forceinline__ v8f wmma_f16(v16h a, v16h b, v8f c) {
  // (neg_a, A, neg_b, B, c_mod, C, reuse_a, reuse_b)
  return __builtin_amdgcn_wmma_f32_16x16x32_f16(false, a, false, b, (short)0, c,
                                                false, false);
}

// A-layout tile (16x32 f16): lane L -> row m=L%16; low VGPRs K=g*8.., high 16+g*8..
__device__ __forceinline__ v16h loadA16(const _Float16* __restrict__ arow,
                                        int kk, int g) {
  return join16(*(const v8h*)(arow + kk + g * 8),
                *(const v8h*)(arow + kk + 16 + g * 8));
}

// B-layout tile (32x16 f16): lane L -> column; low VGPRs K=g*8.., high 16+g*8..
__device__ __forceinline__ v16h loadB16(const _Float16* __restrict__ brow,
                                        int kk, int g) {
  const _Float16* p = brow + kk + g * 8;
  return join16(*(const v8h*)p, *(const v8h*)(p + 16));
}

// ---------------------------------------------------------------------------
// f32 -> f16 cast (4-wide)
// ---------------------------------------------------------------------------
__global__ void cast_f32_f16_v4(const float4* __restrict__ src,
                                v4h* __restrict__ dst, int n4) {
  int i = blockIdx.x * blockDim.x + threadIdx.x;
  int stride = gridDim.x * blockDim.x;
  for (; i < n4; i += stride) {
    float4 f = src[i];
    v4h h;
    h[0] = (_Float16)f.x; h[1] = (_Float16)f.y;
    h[2] = (_Float16)f.z; h[3] = (_Float16)f.w;
    dst[i] = h;
  }
}

// ---------------------------------------------------------------------------
// WMMA GEMM:  out[s][e] = sum_d A16[s][d] * W16[e][d]
// One wave computes a 16x64 output tile (A reused over 4 N-tiles).
// k-loop is double-buffered: loads for step k+32 issue before WMMAs of step k.
//   mode 0 : dst16 = [h][s][dk] f16   (Q, K)
//   mode 1 : dst16 = [e][s]     f16   (V transposed; packed B128 store)
//   mode 2 : dst32 = [s][e]     f32   (final output projection)
// ---------------------------------------------------------------------------
__global__ __launch_bounds__(128)
void gemm16_wmma(const _Float16* __restrict__ A16,
                 const _Float16* __restrict__ W16,
                 _Float16* __restrict__ dst16,
                 float* __restrict__ dst32,
                 int mode) {
  const int lane = threadIdx.x;            // 0..31 (wave32)
  const int wave = threadIdx.y;            // 0..3
  const int job  = blockIdx.x * 4 + wave;  // 0..4095
  const int rowTile = job & 255;           // s0/16
  const int colGrp  = job >> 8;            // 0..15 -> e0 = 64*colGrp
  const int s0 = rowTile * 16;
  const int e0 = colGrp * 64;
  const int m = lane & 15, g = lane >> 4;

  v8f acc[4];
#pragma unroll
  for (int t = 0; t < 4; ++t) acc[t] = vzero8();

  const _Float16* arow = A16 + (s0 + m) * DMODEL;
  const _Float16* brow[4];
#pragma unroll
  for (int t = 0; t < 4; ++t) brow[t] = W16 + (e0 + t * 16 + m) * DMODEL;

  // prologue: buffer 0 holds k=0
  v16h a0 = loadA16(arow, 0, g);
  v16h b0[4];
#pragma unroll
  for (int t = 0; t < 4; ++t) b0[t] = loadB16(brow[t], 0, g);

  for (int kk = 0; kk < DMODEL; kk += 64) {
    // issue loads for k+32 while buffer 0 computes
    v16h a1 = loadA16(arow, kk + 32, g);
    v16h b1[4];
#pragma unroll
    for (int t = 0; t < 4; ++t) b1[t] = loadB16(brow[t], kk + 32, g);
    __builtin_prefetch(arow + kk + 128, 0, 1);
#pragma unroll
    for (int t = 0; t < 4; ++t) acc[t] = wmma_f16(a0, b0[t], acc[t]);

    // issue loads for k+64 (wraps to 0 on last iter; results unused)
    const int kn = (kk + 64) & (DMODEL - 1);
    a0 = loadA16(arow, kn, g);
#pragma unroll
    for (int t = 0; t < 4; ++t) b0[t] = loadB16(brow[t], kn, g);
#pragma unroll
    for (int t = 0; t < 4; ++t) acc[t] = wmma_f16(a1, b1[t], acc[t]);
  }

  // D layout: VGPR r, lanes 0-15 -> (M=r, N=lane); lanes 16-31 -> (M=r+8).
  if (mode == 0) {
    const int h = colGrp;  // e0 = 64*h, tile spans one head
#pragma unroll
    for (int t = 0; t < 4; ++t) {
      const int d = t * 16 + m;
#pragma unroll
      for (int r = 0; r < 8; ++r) {
        const int s = s0 + r + 8 * g;
        dst16[(h * SEQ + s) * DK + d] = (_Float16)acc[t][r];
      }
    }
  } else if (mode == 1) {
#pragma unroll
    for (int t = 0; t < 4; ++t) {
      const int e = e0 + t * 16 + m;
      v8h h8;
#pragma unroll
      for (int r = 0; r < 8; ++r) h8[r] = (_Float16)acc[t][r];
      *(v8h*)(dst16 + e * SEQ + s0 + 8 * g) = h8;  // 8 consecutive s: one B128
    }
  } else {
#pragma unroll
    for (int t = 0; t < 4; ++t) {
      const int e = e0 + t * 16 + m;
#pragma unroll
      for (int r = 0; r < 8; ++r)
        dst32[(s0 + r + 8 * g) * DMODEL + e] = acc[t][r];
    }
  }
}

// ---------------------------------------------------------------------------
// RoPE on Q (with 1/sqrt(dk)=0.125 folded in) and K, in-place, f16.
// One thread per (h, s, pair i<32).
// ---------------------------------------------------------------------------
__global__ void rope_qk(_Float16* __restrict__ Q, _Float16* __restrict__ K,
                        const int* __restrict__ pos) {
  int idx = blockIdx.x * blockDim.x + threadIdx.x;
  if (idx >= NHEADS * SEQ * (DK / 2)) return;
  const int i = idx & 31;
  const int s = (idx >> 5) & (SEQ - 1);
  const int h = idx >> 17;
  const float invf = __expf(-((2.0f * (float)i) / (float)DK) * __logf(10000.0f));
  const float ang = (float)pos[s] * invf;
  float sn, cs;
  __sincosf(ang, &sn, &cs);
  const int base = (h * SEQ + s) * DK + 2 * i;
  const float q1 = (float)Q[base], q2 = (float)Q[base + 1];
  Q[base]     = (_Float16)((q1 * cs - q2 * sn) * 0.125f);
  Q[base + 1] = (_Float16)((q1 * sn + q2 * cs) * 0.125f);
  const float k1 = (float)K[base], k2 = (float)K[base + 1];
  K[base]     = (_Float16)(k1 * cs - k2 * sn);
  K[base + 1] = (_Float16)(k1 * sn + k2 * cs);
}

// ---------------------------------------------------------------------------
// Flash attention: one wave per (head, 16-query block). 32-key chunks.
// Scores: 4 WMMA; PV: 4 WMMA against V^T. Online softmax via half-wave
// __shfl_xor butterflies (C/D layout rows live in 16-lane halves).
// P is moved D-layout -> A-layout through a 1KB LDS tile.
// V B-tiles are loaded immediately after the K tiles so they stay in flight
// across the softmax VALU + LDS phase (partial s_wait_loadcnt).
// Only the final chunk can be partially masked (proved from s0=16*qb).
// ---------------------------------------------------------------------------
__global__ __launch_bounds__(32)
void flash_attn(const _Float16* __restrict__ Qh, const _Float16* __restrict__ Kh,
                const _Float16* __restrict__ Vt, _Float16* __restrict__ AO) {
  __shared__ _Float16 pbuf[16 * 32];
  const int qb = blockIdx.x & 255;
  const int h  = blockIdx.x >> 8;
  const int s0 = qb * 16;
  const int lane = threadIdx.x;
  const int m = lane & 15, g = lane >> 4;

  // Q tile: two 16x32 A-matrices along d
  v16h qa[2];
#pragma unroll
  for (int t = 0; t < 2; ++t) {
    const _Float16* qp = Qh + (h * SEQ + s0 + m) * DK + t * 32;
    qa[t] = join16(*(const v8h*)(qp + g * 8), *(const v8h*)(qp + 16 + g * 8));
  }

  v8f o[4];
#pragma unroll
  for (int t = 0; t < 4; ++t) o[t] = vzero8();
  float mrow[8], lrow[8];
#pragma unroll
  for (int r = 0; r < 8; ++r) { mrow[r] = NEG_BIG; lrow[r] = 0.0f; }

  const int nch = qb / 2 + 1;
  for (int c = 0; c < nch; ++c) {
    const int kb = c * 32;

    // ---- K tiles (consumed by score WMMAs right away) ----
    v16h kb0[2], kb1[2];
#pragma unroll
    for (int t = 0; t < 2; ++t) {
      const _Float16* kp0 = Kh + (h * SEQ + kb + m) * DK + t * 32 + g * 8;
      const _Float16* kp1 = Kh + (h * SEQ + kb + 16 + m) * DK + t * 32 + g * 8;
      kb0[t] = join16(*(const v8h*)kp0, *(const v8h*)(kp0 + 16));
      kb1[t] = join16(*(const v8h*)kp1, *(const v8h*)(kp1 + 16));
    }
    // ---- V tiles: issue now, consumed only after softmax + LDS phase ----
    v16h vb[4];
#pragma unroll
    for (int t = 0; t < 4; ++t) {
      const _Float16* vp = Vt + (h * DK + t * 16 + m) * SEQ + kb + g * 8;
      vb[t] = join16(*(const v8h*)vp, *(const v8h*)(vp + 16));
    }
    // prefetch next chunk's K rows
    __builtin_prefetch(Kh + (h * SEQ + kb + 32 + m) * DK, 0, 1);
    __builtin_prefetch(Kh + (h * SEQ + kb + 48 + m) * DK, 0, 1);

    v8f sc0 = vzero8(), sc1 = vzero8();
#pragma unroll
    for (int t = 0; t < 2; ++t) {
      sc0 = wmma_f16(qa[t], kb0[t], sc0);
      sc1 = wmma_f16(qa[t], kb1[t], sc1);
    }

    const bool diag = (c == nch - 1);
#pragma unroll
    for (int r = 0; r < 8; ++r) {
      const int mg = s0 + r + 8 * g;  // global query row for this VGPR/half
      float v0 = sc0[r], v1 = sc1[r];
      if (diag) {
        if (kb + m > mg)      v0 = NEG_BIG;
        if (kb + 16 + m > mg) v1 = NEG_BIG;
      }
      float loc = fmaxf(v0, v1);
#pragma unroll
      for (int w = 8; w >= 1; w >>= 1) loc = fmaxf(loc, __shfl_xor(loc, w, 16));
      const float nm = fmaxf(mrow[r], loc);
      const float al = __expf(mrow[r] - nm);
      const float p0 = __expf(v0 - nm);
      const float p1 = __expf(v1 - nm);
      float sum = p0 + p1;
#pragma unroll
      for (int w = 8; w >= 1; w >>= 1) sum += __shfl_xor(sum, w, 16);
      lrow[r] = lrow[r] * al + sum;
      mrow[r] = nm;
#pragma unroll
      for (int t = 0; t < 4; ++t) o[t][r] *= al;
      pbuf[(r + 8 * g) * 32 + m]      = (_Float16)p0;
      pbuf[(r + 8 * g) * 32 + 16 + m] = (_Float16)p1;
    }
    __syncthreads();
    // Re-read P in A layout (16 x 32 keys)
    v16h pa = join16(*(const v8h*)(pbuf + m * 32 + g * 8),
                     *(const v8h*)(pbuf + m * 32 + 16 + g * 8));
    __syncthreads();
#pragma unroll
    for (int t = 0; t < 4; ++t) o[t] = wmma_f16(pa, vb[t], o[t]);
  }

  // epilogue: normalize rows, write attn output as [s][D] f16
#pragma unroll
  for (int t = 0; t < 4; ++t) {
    const int e = h * DK + t * 16 + m;
#pragma unroll
    for (int r = 0; r < 8; ++r) {
      const int s = s0 + r + 8 * g;
      AO[s * DMODEL + e] = (_Float16)(o[t][r] / lrow[r]);
    }
  }
}

// ---------------------------------------------------------------------------
// Launch
// ---------------------------------------------------------------------------
extern "C" void kernel_launch(void* const* d_in, const int* in_sizes, int n_in,
                              void* d_out, int out_size, void* d_ws, size_t ws_size,
                              hipStream_t stream) {
  const float* x   = (const float*)d_in[0];
  const int*   pos = (const int*)d_in[1];
  const float* Wq  = (const float*)d_in[2];
  const float* Wk  = (const float*)d_in[3];
  const float* Wv  = (const float*)d_in[4];
  const float* Wo  = (const float*)d_in[5];
  float* out = (float*)d_out;

  char* ws = (char*)d_ws;
  const size_t MB = 1u << 20;
  _Float16* xh  = (_Float16*)(ws + 0 * MB);   // [4096][1024]   8 MiB
  _Float16* wqh = (_Float16*)(ws + 8 * MB);   // [1024][1024]   2 MiB
  _Float16* wkh = (_Float16*)(ws + 10 * MB);
  _Float16* wvh = (_Float16*)(ws + 12 * MB);
  _Float16* woh = (_Float16*)(ws + 14 * MB);
  _Float16* Qh  = (_Float16*)(ws + 16 * MB);  // [16][4096][64] 8 MiB
  _Float16* Kh  = (_Float16*)(ws + 24 * MB);  // [16][4096][64] 8 MiB
  _Float16* Vt  = (_Float16*)(ws + 32 * MB);  // [1024][4096]   8 MiB (V^T)
  _Float16* AOh = (_Float16*)(ws + 40 * MB);  // [4096][1024]   8 MiB

  // 1) casts
  cast_f32_f16_v4<<<1024, 256, 0, stream>>>((const float4*)x,  (v4h*)xh,  SEQ * DMODEL / 4);
  cast_f32_f16_v4<<<512, 256, 0, stream>>>((const float4*)Wq, (v4h*)wqh, DMODEL * DMODEL / 4);
  cast_f32_f16_v4<<<512, 256, 0, stream>>>((const float4*)Wk, (v4h*)wkh, DMODEL * DMODEL / 4);
  cast_f32_f16_v4<<<512, 256, 0, stream>>>((const float4*)Wv, (v4h*)wvh, DMODEL * DMODEL / 4);
  cast_f32_f16_v4<<<512, 256, 0, stream>>>((const float4*)Wo, (v4h*)woh, DMODEL * DMODEL / 4);

  // 2) QKV projections (4096 tile-jobs each, 4 waves/block)
  dim3 gthr(32, 4, 1);
  gemm16_wmma<<<1024, gthr, 0, stream>>>(xh, wqh, Qh, nullptr, 0);
  gemm16_wmma<<<1024, gthr, 0, stream>>>(xh, wkh, Kh, nullptr, 0);
  gemm16_wmma<<<1024, gthr, 0, stream>>>(xh, wvh, Vt, nullptr, 1);

  // 3) RoPE (+ Q scale)
  rope_qk<<<(NHEADS * SEQ * (DK / 2) + 255) / 256, 256, 0, stream>>>(Qh, Kh, pos);

  // 4) attention: 16 heads x 256 query blocks, one wave each
  flash_attn<<<NHEADS * 256, 32, 0, stream>>>(Qh, Kh, Vt, AOh);

  // 5) output projection -> f32 d_out
  gemm16_wmma<<<1024, gthr, 0, stream>>>(AOh, woh, nullptr, out, 2);
}
